// MultiHeadAttention_7206955123484
// MI455X (gfx1250) — matrix operand order repaired and checked by reading, not compiled, not run on hardware
//
#include <hip/hip_runtime.h>

// ---------------------------------------------------------------------------
// MHA forward for MI455X (gfx1250), bf16 WMMA pipeline, v3:
//   - GEMM kernels: 16x64 strip per wave, explicit double-buffered fragment
//     pipeline (loads for k+32 issued before WMMAs of k) -> partial loadcnt
//   - attention: 32 queries/wave; K frags loaded at loop bottom (consumed at
//     next top), V + next-K loads issued before softmax VALU section
//   - flash-style online softmax on S^T (lane == query -> per-lane stats)
// ---------------------------------------------------------------------------

#define NTOK   2048
#define DMODEL 1024
#define HHEADS 16
#define DK     64
#define DV     64

typedef __attribute__((ext_vector_type(16))) __bf16   v16bf;
typedef __attribute__((ext_vector_type(8)))  float    v8f;
typedef __attribute__((ext_vector_type(4)))  unsigned uint4v;  // POD 128-bit

union Frag {
  v16bf          v;
  uint4v         q[2];
  unsigned short us[16];
};

__device__ __forceinline__ unsigned short f32_bf16(float x) {
  unsigned u = __float_as_uint(x);
  u += 0x7FFFu + ((u >> 16) & 1u);   // round-to-nearest-even
  return (unsigned short)(u >> 16);
}

// A-matrix 16x32 bf16 fragment (ISA 7.12.2): lane M=l16,
// elements i -> K = i + 16*half (contiguous, matches B fragments)
__device__ __forceinline__ v16bf load_frag_a(const unsigned short* p, int ld,
                                             int l16, int half) {
  Frag f;
  const unsigned short* r = p + (size_t)l16 * ld;
  f.q[0] = *reinterpret_cast<const uint4v*>(r + 16 * half);
  f.q[1] = *reinterpret_cast<const uint4v*>(r + 16 * half + 8);
  return f.v;
}

// B-matrix 32x16 bf16 fragment: lane N=l16, elements i -> K = i + 16*half.
// Source must hold K contiguous per output column (i.e. transposed weights).
__device__ __forceinline__ v16bf load_frag_b(const unsigned short* p, int ld,
                                             int l16, int half) {
  Frag f;
  const unsigned short* r = p + (size_t)l16 * ld + 16 * half;
  f.q[0] = *reinterpret_cast<const uint4v*>(r);
  f.q[1] = *reinterpret_cast<const uint4v*>(r + 8);
  return f.v;
}

#define WMMA_BF16(a, b, c) \
  __builtin_amdgcn_wmma_f32_16x16x32_bf16(false, (a), false, (b), (short)0, (c), false, false)

// ---------------------------------------------------------------- converters
__global__ void cvt_bf16_kernel(const float* __restrict__ in,
                                unsigned short* __restrict__ out, int n) {
  int i = blockIdx.x * blockDim.x + threadIdx.x;
  if (i < n) out[i] = f32_bf16(in[i]);
}

// (H, 1024, C) f32 -> (H, C, 1024) bf16; C = 1<<logC (power of two dims)
__global__ void cvt_t_kernel(const float* __restrict__ in,
                             unsigned short* __restrict__ out,
                             int logC, int total) {
  int i = blockIdx.x * blockDim.x + threadIdx.x;
  if (i >= total) return;
  int logPer = 10 + logC;                 // per = 1024 * C
  int h = i >> logPer;
  int r = i & ((1 << logPer) - 1);
  int d = r >> logC;
  int c = r & ((1 << logC) - 1);
  out[((size_t)h << logPer) + ((size_t)c << 10) + d] = f32_bf16(in[i]);
}

// ------------------------------------------------------------- projections
// One wave: head h, 16 tokens, FULL head dim (4 n-tiles). A fragment reused
// 4x; all fragments double-buffered so next-k loads overlap current WMMAs.
__global__ __attribute__((amdgpu_waves_per_eu(1)))
void proj_kernel(const unsigned short* __restrict__ Xbf,  // (tok,1024)
                 const unsigned short* __restrict__ WT,   // (H,64,1024)
                 const float* __restrict__ bias,          // (H,64)
                 unsigned short* __restrict__ out,
                 int tokens, int transposeOut) {
  int wid  = blockIdx.x * (blockDim.x >> 5) + (threadIdx.x >> 5);
  int lane = threadIdx.x & 31, l16 = lane & 15, half = lane >> 4;
  int mt = tokens >> 4;
  int h  = wid / mt;
  int m0 = (wid % mt) << 4;

  const unsigned short* xp = Xbf + (size_t)m0 * DMODEL;
  const unsigned short* wt = WT + (size_t)h * DK * DMODEL;

  v8f acc0 = {}, acc1 = {}, acc2 = {}, acc3 = {};
  // prologue: fragments for k0 = 0
  v16bf a  = load_frag_a(xp, DMODEL, l16, half);
  v16bf b0 = load_frag_b(wt,               DMODEL, l16, half);
  v16bf b1 = load_frag_b(wt + 16 * DMODEL, DMODEL, l16, half);
  v16bf b2 = load_frag_b(wt + 32 * DMODEL, DMODEL, l16, half);
  v16bf b3 = load_frag_b(wt + 48 * DMODEL, DMODEL, l16, half);
  for (int k0 = 0; k0 < DMODEL; k0 += 32) {
    int kn = (k0 + 32 < DMODEL) ? k0 + 32 : k0;   // clamp (last iter redundant)
    v16bf an  = load_frag_a(xp + kn, DMODEL, l16, half);
    v16bf bn0 = load_frag_b(wt + kn,               DMODEL, l16, half);
    v16bf bn1 = load_frag_b(wt + 16 * DMODEL + kn, DMODEL, l16, half);
    v16bf bn2 = load_frag_b(wt + 32 * DMODEL + kn, DMODEL, l16, half);
    v16bf bn3 = load_frag_b(wt + 48 * DMODEL + kn, DMODEL, l16, half);
    acc0 = WMMA_BF16(a, b0, acc0);
    acc1 = WMMA_BF16(a, b1, acc1);
    acc2 = WMMA_BF16(a, b2, acc2);
    acc3 = WMMA_BF16(a, b3, acc3);
    a = an; b0 = bn0; b1 = bn1; b2 = bn2; b3 = bn3;
  }
#pragma unroll
  for (int t = 0; t < 4; ++t) {
    v8f acc = (t == 0) ? acc0 : (t == 1) ? acc1 : (t == 2) ? acc2 : acc3;
    int n0 = t << 4;
    float bval = bias[(size_t)h * DK + n0 + l16];
#pragma unroll
    for (int v = 0; v < 8; ++v) {
      int M = v + 8 * half;
      unsigned short w = f32_bf16(acc[v] + bval);
      if (transposeOut)   // (H, 64, tokens) — for V^T A-fragments in attention
        out[(size_t)h * DK * tokens + (size_t)(n0 + l16) * tokens + m0 + M] = w;
      else                // (H, tokens, 64)
        out[(size_t)h * tokens * DK + (size_t)(m0 + M) * DK + n0 + l16] = w;
    }
  }
}

// --------------------------------------------------------- softmax helper
// Transposed online softmax for one 32-key step of one 16-query tile.
// st0/st1 = S^T C-layout tiles (keys j..j+15 / j+16..j+31); lane = query.
// Builds P^T as B-operand fragment pf: element i <-> key (i + 16*half).
__device__ __forceinline__ void softmax_step(const v8f& st0, const v8f& st1,
                                             float& m_run, float& l_run,
                                             v8f& a0, v8f& a1, v8f& a2, v8f& a3,
                                             int half, Frag& pf) {
  float mloc = st0[0];
#pragma unroll
  for (int v = 1; v < 8; ++v) mloc = fmaxf(mloc, st0[v]);
#pragma unroll
  for (int v = 0; v < 8; ++v) mloc = fmaxf(mloc, st1[v]);
  mloc = fmaxf(mloc, __shfl_xor(mloc, 16, 32));
  float m_new = fmaxf(m_run, mloc);
  float corr  = __expf(m_run - m_new);

  float p0[8], p1[8], s = 0.f;
#pragma unroll
  for (int v = 0; v < 8; ++v) {
    p0[v] = __expf(st0[v] - m_new);
    p1[v] = __expf(st1[v] - m_new);
    s += p0[v] + p1[v];
  }
  s += __shfl_xor(s, 16, 32);
  l_run = l_run * corr + s;
  m_run = m_new;
#pragma unroll
  for (int v = 0; v < 8; ++v) { a0[v] *= corr; a1[v] *= corr; a2[v] *= corr; a3[v] *= corr; }
#pragma unroll
  for (int v = 0; v < 8; ++v) {
    float ex0 = __shfl_xor(p0[v], 16, 32);
    float ex1 = __shfl_xor(p1[v], 16, 32);
    pf.us[v]     = f32_bf16(half ? ex1 : p0[v]);  // keys 0..7  / 16..23
    pf.us[v + 8] = f32_bf16(half ? p1[v] : ex0);  // keys 8..15 / 24..31
  }
}

__device__ __forceinline__ void store_o(unsigned short* heads, int q0, int h,
                                        int l16, int half, float l_run,
                                        const v8f& a0, const v8f& a1,
                                        const v8f& a2, const v8f& a3) {
  float sc = 0.125f / l_run;   // reference applies d_k^-0.5 AFTER softmax
  unsigned short* hp = heads + (size_t)(q0 + l16) * (HHEADS * DV) + (size_t)h * DV;
#pragma unroll
  for (int v = 0; v < 8; ++v) {
    int dv = v + 8 * half;
    hp[dv]      = f32_bf16(a0[v] * sc);
    hp[16 + dv] = f32_bf16(a1[v] * sc);
    hp[32 + dv] = f32_bf16(a2[v] * sc);
    hp[48 + dv] = f32_bf16(a3[v] * sc);
  }
}

// ---------------------------------------------------------------- attention
// One wave owns TWO 16-query tiles (32 queries) of one head.
// Pipeline per 32-key step:
//   score WMMAs (K frags loaded at previous bottom) -> issue V loads ->
//   issue next K loads -> softmax VALU (hides loads) -> PV WMMAs
__global__ __attribute__((amdgpu_waves_per_eu(1)))
void attn_kernel(const unsigned short* __restrict__ Qh,   // (H,N,64)
                 const unsigned short* __restrict__ Kh,   // (H,M,64)
                 const unsigned short* __restrict__ VhT,  // (H,64,M)
                 unsigned short* __restrict__ heads,      // (N,H*64)
                 int nq, int nk) {
  int wid  = blockIdx.x * (blockDim.x >> 5) + (threadIdx.x >> 5);
  int lane = threadIdx.x & 31, l16 = lane & 15, half = lane >> 4;
  int qpairs = nq >> 5;
  int h  = wid / qpairs;
  int q0 = (wid % qpairs) << 5;           // 32 queries per wave

  const unsigned short* qa = Qh + (size_t)h * nq * DK + (size_t)q0 * DK;
  const unsigned short* qb = qa + 16 * DK;
  const unsigned short* kh = Kh + (size_t)h * nk * DK;
  const unsigned short* vt = VhT + (size_t)h * DV * nk;

  // Q^T B-operand fragments for both query tiles (d_k = 64 -> 2 frags each)
  v16bf bqa_lo = load_frag_b(qa,      DK, l16, half);
  v16bf bqa_hi = load_frag_b(qa + 32, DK, l16, half);
  v16bf bqb_lo = load_frag_b(qb,      DK, l16, half);
  v16bf bqb_hi = load_frag_b(qb + 32, DK, l16, half);

  v8f accA0 = {}, accA1 = {}, accA2 = {}, accA3 = {};
  v8f accB0 = {}, accB1 = {}, accB2 = {}, accB3 = {};
  v8f zero = {};
  float mA = -1e30f, lA = 0.f, mB = -1e30f, lB = 0.f;

  // prologue: K fragments for j = 0
  v16bf a0l = load_frag_a(kh,                DK, l16, half);
  v16bf a0h = load_frag_a(kh + 32,           DK, l16, half);
  v16bf a1l = load_frag_a(kh + 16 * DK,      DK, l16, half);
  v16bf a1h = load_frag_a(kh + 16 * DK + 32, DK, l16, half);

  for (int j = 0; j < nk; j += 32) {
    // score WMMAs (S^T); K fragments were issued one full iteration ago
    v8f stA0 = WMMA_BF16(a0l, bqa_lo, zero); stA0 = WMMA_BF16(a0h, bqa_hi, stA0);
    v8f stA1 = WMMA_BF16(a1l, bqa_lo, zero); stA1 = WMMA_BF16(a1h, bqa_hi, stA1);
    v8f stB0 = WMMA_BF16(a0l, bqb_lo, zero); stB0 = WMMA_BF16(a0h, bqb_hi, stB0);
    v8f stB1 = WMMA_BF16(a1l, bqb_lo, zero); stB1 = WMMA_BF16(a1h, bqb_hi, stB1);

    // issue V loads for this step (consumed after softmax)
    const unsigned short* vp = vt + j;
    v16bf vf0 = load_frag_a(vp,           nk, l16, half);
    v16bf vf1 = load_frag_a(vp + 16 * nk, nk, l16, half);
    v16bf vf2 = load_frag_a(vp + 32 * nk, nk, l16, half);
    v16bf vf3 = load_frag_a(vp + 48 * nk, nk, l16, half);

    // issue next-iteration K loads (WAR on the score WMMAs above)
    int jn = (j + 32 < nk) ? j + 32 : j;  // clamp (last iter redundant)
    const unsigned short* kpn = kh + (size_t)jn * DK;
    a0l = load_frag_a(kpn,                DK, l16, half);
    a0h = load_frag_a(kpn + 32,           DK, l16, half);
    a1l = load_frag_a(kpn + 16 * DK,      DK, l16, half);
    a1h = load_frag_a(kpn + 16 * DK + 32, DK, l16, half);

    // softmax VALU section hides the outstanding loads
    Frag pfA, pfB;
    softmax_step(stA0, stA1, mA, lA, accA0, accA1, accA2, accA3, half, pfA);
    softmax_step(stB0, stB1, mB, lB, accB0, accB1, accB2, accB3, half, pfB);

    // PV WMMAs; each V fragment reused by both query tiles
    accA0 = WMMA_BF16(vf0, pfA.v, accA0);  accB0 = WMMA_BF16(vf0, pfB.v, accB0);
    accA1 = WMMA_BF16(vf1, pfA.v, accA1);  accB1 = WMMA_BF16(vf1, pfB.v, accB1);
    accA2 = WMMA_BF16(vf2, pfA.v, accA2);  accB2 = WMMA_BF16(vf2, pfB.v, accB2);
    accA3 = WMMA_BF16(vf3, pfA.v, accA3);  accB3 = WMMA_BF16(vf3, pfB.v, accB3);
  }

  store_o(heads, q0,      h, l16, half, lA, accA0, accA1, accA2, accA3);
  store_o(heads, q0 + 16, h, l16, half, lB, accB0, accB1, accB2, accB3);
}

// ---------------------------------------------------------- output GEMM
// One wave computes a 16x64 strip of out = heads @ Wo + bo; same
// double-buffered pipeline as proj_kernel.
__global__ __attribute__((amdgpu_waves_per_eu(1)))
void out_gemm_kernel(const unsigned short* __restrict__ A,   // (N,1024)
                     const unsigned short* __restrict__ WoT, // (1024,1024)
                     const float* __restrict__ bo,
                     float* __restrict__ out) {
  int wid  = blockIdx.x * (blockDim.x >> 5) + (threadIdx.x >> 5);
  int lane = threadIdx.x & 31, l16 = lane & 15, half = lane >> 4;
  int nstrips = DMODEL >> 6;            // 16 strips of 64 cols
  int m0 = (wid / nstrips) << 4;
  int n0 = (wid % nstrips) << 6;

  const unsigned short* ap = A + (size_t)m0 * DMODEL;
  const unsigned short* wp = WoT + (size_t)n0 * DMODEL;

  v8f acc0 = {}, acc1 = {}, acc2 = {}, acc3 = {};
  v16bf a  = load_frag_a(ap, DMODEL, l16, half);
  v16bf b0 = load_frag_b(wp,               DMODEL, l16, half);
  v16bf b1 = load_frag_b(wp + 16 * DMODEL, DMODEL, l16, half);
  v16bf b2 = load_frag_b(wp + 32 * DMODEL, DMODEL, l16, half);
  v16bf b3 = load_frag_b(wp + 48 * DMODEL, DMODEL, l16, half);
  for (int k0 = 0; k0 < DMODEL; k0 += 32) {
    int kn = (k0 + 32 < DMODEL) ? k0 + 32 : k0;
    v16bf an  = load_frag_a(ap + kn, DMODEL, l16, half);
    v16bf bn0 = load_frag_b(wp + kn,               DMODEL, l16, half);
    v16bf bn1 = load_frag_b(wp + 16 * DMODEL + kn, DMODEL, l16, half);
    v16bf bn2 = load_frag_b(wp + 32 * DMODEL + kn, DMODEL, l16, half);
    v16bf bn3 = load_frag_b(wp + 48 * DMODEL + kn, DMODEL, l16, half);
    acc0 = WMMA_BF16(a, b0, acc0);
    acc1 = WMMA_BF16(a, b1, acc1);
    acc2 = WMMA_BF16(a, b2, acc2);
    acc3 = WMMA_BF16(a, b3, acc3);
    a = an; b0 = bn0; b1 = bn1; b2 = bn2; b3 = bn3;
  }
#pragma unroll
  for (int t = 0; t < 4; ++t) {
    v8f acc = (t == 0) ? acc0 : (t == 1) ? acc1 : (t == 2) ? acc2 : acc3;
    int nn = n0 + (t << 4);
    float bv = bo[nn + l16];
#pragma unroll
    for (int v = 0; v < 8; ++v) {
      int M = m0 + v + 8 * half;
      out[(size_t)M * DMODEL + nn + l16] = acc[v] + bv;
    }
  }
}

// ----------------------------------------------------------------- launcher
extern "C" void kernel_launch(void* const* d_in, const int* in_sizes, int n_in,
                              void* d_out, int out_size, void* d_ws, size_t ws_size,
                              hipStream_t stream) {
  const float* Q  = (const float*)d_in[0];
  const float* K  = (const float*)d_in[1];
  const float* V  = (const float*)d_in[2];
  // d_in[3] = mask (unused by the reference forward)
  const float* Wq = (const float*)d_in[4];
  const float* bq = (const float*)d_in[5];
  const float* Wk = (const float*)d_in[6];
  const float* bk = (const float*)d_in[7];
  const float* Wv = (const float*)d_in[8];
  const float* bv = (const float*)d_in[9];
  const float* Wo = (const float*)d_in[10];
  const float* bo = (const float*)d_in[11];
  float* out = (float*)d_out;

  char* ws = (char*)d_ws;
  size_t off = 0;
  auto alloc = [&](size_t bytes) -> unsigned short* {
    unsigned short* p = (unsigned short*)(ws + off);
    off = (off + bytes + 255) & ~(size_t)255;
    return p;
  };

  const size_t tokBytes = (size_t)NTOK * DMODEL * 2;          // 4 MB
  unsigned short* Qbf  = alloc(tokBytes);
  unsigned short* Kbf  = alloc(tokBytes);
  unsigned short* Vbf  = alloc(tokBytes);
  unsigned short* WqT  = alloc((size_t)HHEADS * DK * DMODEL * 2);
  unsigned short* WkT  = alloc((size_t)HHEADS * DK * DMODEL * 2);
  unsigned short* WvT  = alloc((size_t)HHEADS * DV * DMODEL * 2);
  unsigned short* WoT  = alloc((size_t)DMODEL * DMODEL * 2);
  unsigned short* Qh   = alloc((size_t)HHEADS * NTOK * DK * 2);
  unsigned short* Khb  = alloc((size_t)HHEADS * NTOK * DK * 2);
  unsigned short* VhT  = alloc((size_t)HHEADS * DV * NTOK * 2);
  unsigned short* Hd   = alloc((size_t)NTOK * HHEADS * DV * 2);
  (void)ws_size; (void)in_sizes; (void)n_in; (void)out_size;

  const int TB = 256;
  // 1) bf16 conversions (+ weight transposes; all dims are powers of two)
  int ntok = NTOK * DMODEL;
  cvt_bf16_kernel<<<(ntok + TB - 1) / TB, TB, 0, stream>>>(Q, Qbf, ntok);
  cvt_bf16_kernel<<<(ntok + TB - 1) / TB, TB, 0, stream>>>(K, Kbf, ntok);
  cvt_bf16_kernel<<<(ntok + TB - 1) / TB, TB, 0, stream>>>(V, Vbf, ntok);
  int nw = HHEADS * DMODEL * DK;
  cvt_t_kernel<<<(nw + TB - 1) / TB, TB, 0, stream>>>(Wq, WqT, 6, nw);
  cvt_t_kernel<<<(nw + TB - 1) / TB, TB, 0, stream>>>(Wk, WkT, 6, nw);
  cvt_t_kernel<<<(nw + TB - 1) / TB, TB, 0, stream>>>(Wv, WvT, 6, nw);
  int nwo = DMODEL * DMODEL;
  cvt_t_kernel<<<(nwo + TB - 1) / TB, TB, 0, stream>>>(Wo, WoT, 10, nwo);

  // 2) projections: H*(2048/16) = 2048 waves (each does full 64-wide head)
  int projBlocks = (HHEADS * (NTOK / 16)) / 8;
  proj_kernel<<<projBlocks, TB, 0, stream>>>(Qbf, WqT, bq, Qh,  NTOK, 0);
  proj_kernel<<<projBlocks, TB, 0, stream>>>(Kbf, WkT, bk, Khb, NTOK, 0);
  proj_kernel<<<projBlocks, TB, 0, stream>>>(Vbf, WvT, bv, VhT, NTOK, 1);

  // 3) flash attention: H*(2048/32) = 1024 waves (32 queries each)
  int attnBlocks = (HHEADS * (NTOK / 32)) / 8;
  attn_kernel<<<attnBlocks, TB, 0, stream>>>(Qh, Khb, VhT, Hd, NTOK, NTOK);

  // 4) output projection: (2048/16)*(1024/64) = 2048 waves
  int outBlocks = ((NTOK / 16) * (DMODEL / 64)) / 8;
  out_gemm_kernel<<<outBlocks, TB, 0, stream>>>(Hd, WoT, bo, out);
}
